// NASEERLayer_88656714924947
// MI455X (gfx1250) — matrix operand order repaired and checked by reading, not compiled, not run on hardware
//
#include <hip/hip_runtime.h>
#include <hip/hip_fp16.h>
#include <math.h>

typedef __attribute__((ext_vector_type(16))) _Float16 v16h;
typedef __attribute__((ext_vector_type(8)))  float    v8f;

constexpr int NB_B = 4;
constexpr int NT   = 512;   // T
constexpr int ND   = 128;   // D
constexpr int NNB  = 4;     // NB
constexpr int NTOPK = 32;
constexpr int NROWS = NB_B * NT;   // 2048

// ---- dynamic LDS layout for the hot kernel ----
constexpr int INTER_LD = 132;            // padded f32 row stride for interaction
constexpr int WF2_LD   = 136;            // padded f16 row stride for W_f2^T
constexpr size_t OFF_INTER  = 0;
constexpr size_t SZ_INTER   = (size_t)NT * INTER_LD * sizeof(float);     // 270336
constexpr size_t OFF_WF2    = OFF_INTER + SZ_INTER;
constexpr size_t SZ_WF2     = (size_t)ND * WF2_LD * sizeof(_Float16);    // 34816
constexpr size_t OFF_PIB    = OFF_WF2 + SZ_WF2;                          // 305152
constexpr size_t OFF_SCORES = OFF_PIB + ND * sizeof(float);              // 305664
constexpr size_t OFF_KEYS   = OFF_SCORES + NT * sizeof(float);           // 307712 (8B aligned)
constexpr size_t OFF_RED    = OFF_KEYS + NT * sizeof(unsigned long long);// 311808
constexpr size_t OFF_PICK   = OFF_RED + 256 * sizeof(unsigned long long);// 313856
constexpr size_t OFF_XENT   = OFF_PICK + NTOPK * sizeof(int);            // 313984
constexpr size_t OFF_Y      = OFF_XENT + ND * sizeof(float);             // 314496
constexpr size_t SMEM_BYTES = OFF_Y + ND * sizeof(float);                // 315008
static_assert(SMEM_BYTES <= 320u * 1024u, "LDS budget exceeded");
static_assert(OFF_KEYS % 8 == 0 && OFF_RED % 8 == 0, "align");

// ===================================================================
// Kernel 1: x_super = softmax(x@W_g+b_g) basis-mix of (x@W_phi+b_phi)
// ===================================================================
__global__ __launch_bounds__(128) void k_super(
    const float* __restrict__ x,   const float* __restrict__ Wphi,
    const float* __restrict__ bphi,const float* __restrict__ Wg,
    const float* __restrict__ bg,  float* __restrict__ xsup)
{
    const int row = blockIdx.x;
    const int d   = threadIdx.x;
    __shared__ float xs[ND];
    __shared__ float gs[NNB];
    xs[d] = x[(size_t)row * ND + d];
    __syncthreads();

    float a0 = 0.f, a1 = 0.f, a2 = 0.f, a3 = 0.f;
    const float* wp = Wphi + d;
#pragma unroll 4
    for (int k = 0; k < ND; ++k) {
        const float xv = xs[k];
        const float* wk = wp + (size_t)k * (ND * NNB);
        a0 = fmaf(xv, wk[0 * ND], a0);
        a1 = fmaf(xv, wk[1 * ND], a1);
        a2 = fmaf(xv, wk[2 * ND], a2);
        a3 = fmaf(xv, wk[3 * ND], a3);
    }
    if (d < NNB) {
        float g = bg[d];
        for (int k = 0; k < ND; ++k) g = fmaf(xs[k], Wg[k * NNB + d], g);
        gs[d] = g;
    }
    __syncthreads();
    const float g0 = gs[0], g1 = gs[1], g2 = gs[2], g3 = gs[3];
    const float m  = fmaxf(fmaxf(g0, g1), fmaxf(g2, g3));
    const float e0 = expf(g0 - m), e1 = expf(g1 - m);
    const float e2 = expf(g2 - m), e3 = expf(g3 - m);
    const float inv = 1.0f / (e0 + e1 + e2 + e3);
    float r = (a0 + bphi[0 * ND + d]) * e0 + (a1 + bphi[1 * ND + d]) * e1
            + (a2 + bphi[2 * ND + d]) * e2 + (a3 + bphi[3 * ND + d]) * e3;
    xsup[(size_t)row * ND + d] = r * inv;
}

// ===================================================================
// Kernel 2: pi = x_super @ W_f1[:D], pj = x_super @ W_f1[D:]
// ===================================================================
__global__ __launch_bounds__(256) void k_pipj(
    const float* __restrict__ xsup, const float* __restrict__ Wf1,
    float* __restrict__ pi, float* __restrict__ pj)
{
    const int row = blockIdx.x;
    const int tid = threadIdx.x;
    __shared__ float xs[ND];
    if (tid < ND) xs[tid] = xsup[(size_t)row * ND + tid];
    __syncthreads();
    const int d = tid & (ND - 1);
    const float* w = Wf1 + (tid < ND ? 0 : ND * ND) + d;
    float a = 0.f;
#pragma unroll 4
    for (int k = 0; k < ND; ++k) a = fmaf(xs[k], w[(size_t)k * ND], a);
    if (tid < ND) pi[(size_t)row * ND + d] = a;
    else          pj[(size_t)row * ND + d] = a;
}

// ===================================================================
// Kernel 3 (hot): per (b,i): interaction = relu(pi+pj+b_f1)@W_f2+b_f2
// 8 waves = 4 j-groups x 2 dt-groups. Each wave keeps its 16 B-frags
// (4 kc x 4 dt) of W_f2 in 128 VGPRs for the whole block: zero LDS
// B traffic in the GEMM loop. Row norms combined via ds_add_f32
// (exactly 2 deterministic contributions per row), then top-32,
// entangled sum, @W_U, LayerNorm.
// ===================================================================
__global__ __launch_bounds__(256) void k_entangle(
    const float* __restrict__ xsup, const float* __restrict__ pi,
    const float* __restrict__ pj,   const float* __restrict__ bf1,
    const float* __restrict__ Wf2,  const float* __restrict__ bf2,
    const float* __restrict__ WU,   const float* __restrict__ bU,
    const float* __restrict__ gam,  const float* __restrict__ bet,
    float* __restrict__ out)
{
    extern __shared__ char smem[];
    float*              inter  = (float*)(smem + OFF_INTER);
    _Float16*           wf2h   = (_Float16*)(smem + OFF_WF2);
    float*              pib    = (float*)(smem + OFF_PIB);
    float*              scores = (float*)(smem + OFF_SCORES);
    unsigned long long* keys   = (unsigned long long*)(smem + OFF_KEYS);
    unsigned long long* red    = (unsigned long long*)(smem + OFF_RED);
    int*                picked = (int*)(smem + OFF_PICK);
    float*              xent   = (float*)(smem + OFF_XENT);
    float*              yrow   = (float*)(smem + OFF_Y);

    const int row = blockIdx.x;           // b*T + i
    const int tid = threadIdx.x;
    const int b   = row >> 9;             // T = 512

    // stage W_f2 transposed [n][k] as f16 (padded rows -> conflict-free loads)
    for (int idx = tid; idx < ND * ND; idx += 256) {
        const int k = idx >> 7, n = idx & 127;
        wf2h[n * WF2_LD + k] = (_Float16)Wf2[idx];
    }
    if (tid < ND) pib[tid] = pi[(size_t)row * ND + tid] + bf1[tid];
    scores[tid]       = 0.f;
    scores[tid + 256] = 0.f;
    __syncthreads();

    const int wave    = tid >> 5;
    const int jgroup  = wave & 3;         // 0..3
    const int dgroup  = wave >> 2;        // 0..1 -> output cols [dgroup*64, +64)
    const int lane    = tid & 31;
    const int halfSel = lane >> 4;        // 0: lanes 0-15, 1: lanes 16-31
    const int mrow    = lane & 15;
    const float2* pib2   = (const float2*)pib;
    const float*  pjBase = pj + (size_t)b * NT * ND;

    // ---- load this wave's 16 B fragments (4 kc x 4 dt) into registers ----
    v16h bfr[16];
#pragma unroll
    for (int kc = 0; kc < 4; ++kc) {
#pragma unroll
        for (int t = 0; t < 4; ++t) {
            const int dt = dgroup * 4 + t;
            union { v16h h; uint4 u4[2]; } bf;
            const uint4* bp4 = (const uint4*)
                (wf2h + (dt * 16 + mrow) * WF2_LD + kc * 32 + halfSel * 16);
            bf.u4[0] = bp4[0];
            bf.u4[1] = bp4[1];
            bfr[kc * 4 + t] = bf.h;
        }
    }
    float bf2r[4];
#pragma unroll
    for (int t = 0; t < 4; ++t) bf2r[t] = bf2[(dgroup * 4 + t) * 16 + mrow];

    for (int p = 0; p < 8; ++p) {
        const int jt = jgroup + p * 4;             // j-tile 0..31
        const int j  = jt * 16 + mrow;             // A row for this lane
        const float2* pjrow2 = (const float2*)(pjBase + (size_t)j * ND);

        // ---- build A fragments: h = relu(pi+pj+b_f1), k = 16*half + s order, same as B ----
        v16h afrag[4];
#pragma unroll
        for (int kc = 0; kc < 4; ++kc) {
            const int base = kc * 32 + halfSel * 16;
#pragma unroll
            for (int v = 0; v < 8; ++v) {
                const int kk = base + 2 * v;
                const float2 pjv = pjrow2[kk >> 1];
                const float2 piv = pib2[kk >> 1];
                afrag[kc][2 * v + 0] = (_Float16)fmaxf(pjv.x + piv.x, 0.f);
                afrag[kc][2 * v + 1] = (_Float16)fmaxf(pjv.y + piv.y, 0.f);
            }
        }

        // ---- GEMM: 4 kc x 4 dt WMMAs, all B operands already in VGPRs ----
        v8f acc[4] = {};
#pragma unroll
        for (int kc = 0; kc < 4; ++kc) {
#pragma unroll
            for (int t = 0; t < 4; ++t) {
                acc[t] = __builtin_amdgcn_wmma_f32_16x16x32_f16(
                    false, afrag[kc], false, bfr[kc * 4 + t],
                    (short)0, acc[t], false, false);
            }
        }

        // ---- bias, store interaction slice, partial row sq-sums ----
        const int jrow0 = jt * 16 + halfSel * 8;   // VGPR r -> row jrow0+r
        float rowsq[8];
#pragma unroll
        for (int r = 0; r < 8; ++r) rowsq[r] = 0.f;
#pragma unroll
        for (int t = 0; t < 4; ++t) {
            const int col = dgroup * 64 + t * 16 + mrow;
#pragma unroll
            for (int r = 0; r < 8; ++r) {
                const float val = acc[t][r] + bf2r[t];
                inter[(jrow0 + r) * INTER_LD + col] = val;
                rowsq[r] = fmaf(val, val, rowsq[r]);
            }
        }
        // reduce over the 16 columns held by this 16-lane half (64 cols total)
#pragma unroll
        for (int r = 0; r < 8; ++r) {
            float s = rowsq[r];
            s += __shfl_xor(s, 1, 32);
            s += __shfl_xor(s, 2, 32);
            s += __shfl_xor(s, 4, 32);
            s += __shfl_xor(s, 8, 32);
            rowsq[r] = s;
        }
        if (mrow == 0) {   // lanes 0 & 16: add this wave's 64-col partial
#pragma unroll
            for (int r = 0; r < 8; ++r)
                atomicAdd(&scores[jrow0 + r], rowsq[r]);
        }
    }
    __syncthreads();

    // ---- pack sort keys: sqsum >= 0 so float bits order as unsigned;
    //      low bits break ties toward LOWER index (matches jax.lax.top_k) ----
    for (int j = tid; j < NT; j += 256)
        keys[j] = ((unsigned long long)__float_as_uint(scores[j]) << 32)
                | (unsigned int)(511 - j);
    __syncthreads();

    // ---- top-32 by iterative argmax over 512 keys ----
    for (int it = 0; it < NTOPK; ++it) {
        const unsigned long long k1 = keys[tid];
        const unsigned long long k2 = keys[tid + 256];
        red[tid] = (k1 > k2) ? k1 : k2;
        __syncthreads();
        for (int s = 128; s > 0; s >>= 1) {
            if (tid < s) {
                const unsigned long long o = red[tid + s];
                if (o > red[tid]) red[tid] = o;
            }
            __syncthreads();
        }
        if (tid == 0) {
            const unsigned long long top = red[0];
            const int jj = 511 - (int)(top & 0xFFFFFFFFull);
            picked[it] = jj;
            keys[jj] = 0ull;     // remove from candidate set
        }
        __syncthreads();
    }

    // ---- entangled sum + residual ----
    if (tid < ND) {
        float ent = 0.f;
#pragma unroll
        for (int kk = 0; kk < NTOPK; ++kk)
            ent += inter[picked[kk] * INTER_LD + tid];
        xent[tid] = xsup[(size_t)row * ND + tid] + ent;
    }
    __syncthreads();

    // ---- y = x_ent @ W_U + b_U ----
    if (tid < ND) {
        float y = bU[tid];
        const float* w = WU + tid;
#pragma unroll 4
        for (int k = 0; k < ND; ++k) y = fmaf(xent[k], w[(size_t)k * ND], y);
        yrow[tid] = y;
    }
    float* redf = (float*)red;     // reuse as f32 reduction buffer (256 slots)
    __syncthreads();
    redf[tid] = (tid < ND) ? yrow[tid] : 0.f;
    __syncthreads();
    for (int s = 128; s > 0; s >>= 1) {
        if (tid < s) redf[tid] += redf[tid + s];
        __syncthreads();
    }
    const float mu = redf[0] * (1.0f / ND);
    __syncthreads();
    {
        const float dlt = (tid < ND) ? (yrow[tid] - mu) : 0.f;
        redf[tid] = dlt * dlt;
    }
    __syncthreads();
    for (int s = 128; s > 0; s >>= 1) {
        if (tid < s) redf[tid] += redf[tid + s];
        __syncthreads();
    }
    const float var = redf[0] * (1.0f / ND);
    if (tid < ND) {
        const float o = (yrow[tid] - mu) / sqrtf(var + 1e-5f);
        out[(size_t)row * ND + tid] = o * gam[tid] + bet[tid];
    }
}

// ===================================================================
extern "C" void kernel_launch(void* const* d_in, const int* in_sizes, int n_in,
                              void* d_out, int out_size, void* d_ws, size_t ws_size,
                              hipStream_t stream)
{
    const float* x    = (const float*)d_in[0];
    const float* Wphi = (const float*)d_in[1];
    const float* bphi = (const float*)d_in[2];
    const float* Wg   = (const float*)d_in[3];
    const float* bg   = (const float*)d_in[4];
    const float* Wf1  = (const float*)d_in[5];
    const float* bf1  = (const float*)d_in[6];
    const float* Wf2  = (const float*)d_in[7];
    const float* bf2  = (const float*)d_in[8];
    const float* WU   = (const float*)d_in[9];
    const float* bU   = (const float*)d_in[10];
    const float* gam  = (const float*)d_in[11];
    const float* bet  = (const float*)d_in[12];
    float* out = (float*)d_out;

    float* xsup = (float*)d_ws;                 // 2048*128 f32
    float* pi   = xsup + (size_t)NROWS * ND;    // 2048*128 f32
    float* pj   = pi   + (size_t)NROWS * ND;    // 2048*128 f32  (3 MB total)

    k_super<<<NROWS, 128, 0, stream>>>(x, Wphi, bphi, Wg, bg, xsup);
    k_pipj <<<NROWS, 256, 0, stream>>>(xsup, Wf1, pi, pj);

    (void)hipFuncSetAttribute((const void*)k_entangle,
                              hipFuncAttributeMaxDynamicSharedMemorySize,
                              (int)SMEM_BYTES);
    k_entangle<<<NROWS, 256, SMEM_BYTES, stream>>>(
        xsup, pi, pj, bf1, Wf2, bf2, WU, bU, gam, bet, out);
}
